// Agent_global_multistep_22333829940022
// MI455X (gfx1250) — matrix operand order repaired and checked by reading, not compiled, not run on hardware
//
#include <hip/hip_runtime.h>
#include <hip/hip_bf16.h>

// ---------------------------------------------------------------------------
// Model dims
// ---------------------------------------------------------------------------
#define BB    32          // batch
#define SEQV  516         // valid sequence length (256*2 + 4)
#define SPD   544         // padded sequence (multiple of 32)
#define DM    240         // d_model
#define DPAD  256         // padded d_model (multiple of 32)
#define NH    8           // heads
#define DH    30          // head dim
#define DHP   32          // padded head dim
#define FFN   1024
#define BS    (BB * SPD)  // 17408 padded rows
#define MT    (BS / 16)   // 1088 M tiles

typedef __bf16 bf16;
typedef __attribute__((ext_vector_type(8)))  bf16  v8bf;
typedef __attribute__((ext_vector_type(16))) bf16  v16bf;
typedef __attribute__((ext_vector_type(8)))  float v8f;

// ---------------------------------------------------------------------------
// WMMA fragment helpers (CDNA5 wave32 layouts, CDNA5 ISA §7.12.2)
// ---------------------------------------------------------------------------
__device__ __forceinline__ v16bf frag_cat(v8bf lo, v8bf hi) {
  return __builtin_shufflevector(lo, hi, 0,1,2,3,4,5,6,7,8,9,10,11,12,13,14,15);
}

// A operand: 16x32 tile of row-major bf16 [*, lda]; rows row0.., cols kb..kb+31.
// lane<16 holds K {kb..kb+7, kb+16..kb+23}; lane>=16 holds {kb+8..15, kb+24..31}.
__device__ __forceinline__ v16bf load_frag_a(const bf16* A, int lda, int row0, int kb, int lane) {
  int m  = row0 + (lane & 15);
  int kh = (lane & 16) ? 8 : 0;
  const bf16* p = A + (size_t)m * lda + kb + kh;
  v8bf lo = *(const v8bf*)p;
  v8bf hi = *(const v8bf*)(p + 16);
  return frag_cat(lo, hi);
}

// B operand: Bmat[k][n] = R[n0+n][kb+k] with R row-major bf16 [*, ldr].
// lane<16 holds K {kb..kb+7, kb+16..kb+23}; lane>=16 holds {kb+8..15, kb+24..31}.
__device__ __forceinline__ v16bf load_frag_b(const bf16* R, int ldr, int n0, int kb, int lane) {
  int n  = n0 + (lane & 15);
  int ks = kb + ((lane & 16) ? 8 : 0);
  const bf16* p = R + (size_t)n * ldr + ks;
  v8bf lo = *(const v8bf*)p;
  v8bf hi = *(const v8bf*)(p + 16);
  return frag_cat(lo, hi);
}

__device__ __forceinline__ v8f wmma_bf16(v16bf a, v16bf b, v8f c) {
  return __builtin_amdgcn_wmma_f32_16x16x32_bf16(false, a, false, b, (short)0, c, false, false);
}

__device__ __forceinline__ float gelu_f(float x) {
  return 0.5f * x * (1.0f + erff(x * 0.70710678118654752f));
}

__device__ __forceinline__ float wave_sum(float v) {
#pragma unroll
  for (int m = 16; m >= 1; m >>= 1) v += __shfl_xor(v, m, 32);
  return v;
}

// ---------------------------------------------------------------------------
// Weight convert: W[N][K] f32 -> Wb[N][Kpad] bf16, zero padded
// ---------------------------------------------------------------------------
__global__ void wconv_kernel(const float* __restrict__ W, bf16* __restrict__ Wb,
                             int N, int K, int Kpad) {
  int gid = blockIdx.x * blockDim.x + threadIdx.x;
  if (gid >= N * Kpad) return;
  int n = gid / Kpad, k = gid - n * Kpad;
  Wb[gid] = (k < K) ? (bf16)W[(size_t)n * K + k] : (bf16)0.f;
}

// ---------------------------------------------------------------------------
// WMMA GEMM: C[M,N] = act(A[M,Kpad]_bf16 x Wb[N,Kpad]_bf16^T + bias[N])
// Each wave computes MREP 16x16 tiles along M (B frag reused MREP times).
// Multi-wave blocks: wave w handles mtile (blockIdx.x*nwaves + w)*MREP.
// ---------------------------------------------------------------------------
template <int MREP, int ACT, int OUTBF>
__global__ __launch_bounds__(128) void gemm_kernel(
    const bf16* __restrict__ A, int lda,
    const bf16* __restrict__ Wb, int ldw, int Kround,
    const float* __restrict__ bias,
    float* __restrict__ Cf, bf16* __restrict__ Cb, int ldc) {
  int lane = threadIdx.x & 31;
  int wid  = threadIdx.x >> 5;
  int nw   = blockDim.x >> 5;
  int row0 = (blockIdx.x * nw + wid) * (MREP * 16);
  int n0   = blockIdx.y * 16;
  v8f acc[MREP];
#pragma unroll
  for (int i = 0; i < MREP; ++i) acc[i] = (v8f){0.f,0.f,0.f,0.f,0.f,0.f,0.f,0.f};
  for (int kb = 0; kb < Kround; kb += 32) {
    v16bf b = load_frag_b(Wb, ldw, n0, kb, lane);
    if (kb + 32 < Kround)  // prefetch next A chunk (global_prefetch_b8)
      __builtin_prefetch(A + (size_t)(row0 + (lane & 15)) * lda + kb + 32, 0, 1);
#pragma unroll
    for (int i = 0; i < MREP; ++i) {
      v16bf a = load_frag_a(A, lda, row0 + i * 16, kb, lane);
      acc[i] = wmma_bf16(a, b, acc[i]);
    }
  }
  int n  = n0 + (lane & 15);
  int mo = (lane & 16) ? 8 : 0;
  float bv = bias[n];
#pragma unroll
  for (int i = 0; i < MREP; ++i) {
#pragma unroll
    for (int r = 0; r < 8; ++r) {
      float v = acc[i][r] + bv;
      if (ACT == 1) v = gelu_f(v);
      size_t idx = (size_t)(row0 + i * 16 + mo + r) * ldc + n;
      if (OUTBF) Cb[idx] = (bf16)v;
      else       Cf[idx] = v;
    }
  }
}

// ---------------------------------------------------------------------------
// Assemble src tokens + coords:  [B,SPD,240] f32 + [B,SPD,256] bf16 + [B,SPD,3]
// ---------------------------------------------------------------------------
__global__ void assemble_kernel(
    const float* __restrict__ hand, const float* __restrict__ head,
    const float* __restrict__ chand, const float* __restrict__ chead,
    const float* __restrict__ state, const float* __restrict__ text,
    const float* __restrict__ perc, const float* __restrict__ htr,
    const float* __restrict__ headtr,
    const float* __restrict__ me_img, const float* __restrict__ me_3d,
    float* __restrict__ Xres, bf16* __restrict__ Xbf, float* __restrict__ coords) {
  int gid = blockIdx.x * blockDim.x + threadIdx.x;
  if (gid >= BB * SPD) return;
  int s = gid % SPD, b = gid / SPD;
  float* xr = Xres + (size_t)gid * DM;
  bf16*  xb = Xbf  + (size_t)gid * DPAD;
  float c0 = 0.f, c1 = 0.f, c2 = 0.f;
  if (s < SEQV) {
    for (int d = 0; d < DM; ++d) {
      float v;
      if (s == 0)        v = state[b * DM + d];
      else if (s == 1)   v = text[b * DM + d];
      else if (s < 4)    v = perc[((size_t)b * 2 + (s - 2)) * DM + d] + me_3d[d];
      else if (s < 260)  v = hand[((size_t)b * 256 + (s - 4)) * DM + d] + me_img[d];
      else               v = head[((size_t)b * 256 + (s - 260)) * DM + d] + me_img[d];
      xr[d] = v;
      xb[d] = (bf16)v;
    }
    if (s == 2)      { c0 = htr[b*3];    c1 = htr[b*3+1];    c2 = htr[b*3+2]; }
    else if (s == 3) { c0 = headtr[b*3]; c1 = headtr[b*3+1]; c2 = headtr[b*3+2]; }
    else if (s >= 4 && s < 260) {
      const float* cp = chand + ((size_t)b * 256 + (s - 4)) * 3;
      c0 = cp[0]; c1 = cp[1]; c2 = cp[2];
    } else if (s >= 260) {
      const float* cp = chead + ((size_t)b * 256 + (s - 260)) * 3;
      c0 = cp[0]; c1 = cp[1]; c2 = cp[2];
    }
  } else {
    for (int d = 0; d < DM; ++d) { xr[d] = 0.f; xb[d] = (bf16)0.f; }
  }
  for (int d = DM; d < DPAD; ++d) xb[d] = (bf16)0.f;
  coords[(size_t)gid * 3 + 0] = c0;
  coords[(size_t)gid * 3 + 1] = c1;
  coords[(size_t)gid * 3 + 2] = c2;
}

// ---------------------------------------------------------------------------
// Rotary 3D PE + pack into per-head bf16 [B,NH,SPD,DHP]
// ---------------------------------------------------------------------------
__global__ void rope_pack_kernel(const float* __restrict__ T,
                                 const float* __restrict__ coords,
                                 bf16* __restrict__ Dst) {
  int gid = blockIdx.x * blockDim.x + threadIdx.x;
  if (gid >= BB * NH * SPD) return;
  int s = gid % SPD;
  int h = (gid / SPD) % NH;
  int b = gid / (SPD * NH);
  const float* src = T + ((size_t)b * SPD + s) * DM + h * DH;
  const float* c   = coords + ((size_t)b * SPD + s) * 3;
  bf16* dst = Dst + (size_t)gid * DHP;
  const float freqs[5] = {1.0f, 0.15848931924611134f, 0.025118864315095794f,
                          0.003981071705534972f, 0.000630957344480193f};
#pragma unroll
  for (int j = 0; j < 15; ++j) {
    float ang = c[j / 5] * freqs[j % 5];
    float sn = __sinf(ang), cs = __cosf(ang);
    float x1 = src[2 * j], x2 = src[2 * j + 1];
    dst[2 * j]     = (bf16)(x1 * cs - x2 * sn);
    dst[2 * j + 1] = (bf16)(x1 * sn + x2 * cs);
  }
  dst[30] = (bf16)0.f;
  dst[31] = (bf16)0.f;
}

// V pack transposed: Vt[B,NH,DHP,SPD]  (B-operand friendly contiguous-K loads)
__global__ void v_pack_kernel(const float* __restrict__ T, bf16* __restrict__ Vt) {
  int gid = blockIdx.x * blockDim.x + threadIdx.x;
  if (gid >= BB * NH * SPD) return;
  int s = gid % SPD;
  int h = (gid / SPD) % NH;
  int b = gid / (SPD * NH);
  const float* src = T + ((size_t)b * SPD + s) * DM + h * DH;
  bf16* base = Vt + ((size_t)b * NH + h) * (size_t)DHP * SPD;
#pragma unroll
  for (int d = 0; d < DH; ++d) base[(size_t)d * SPD + s] = (bf16)src[d];
  base[(size_t)30 * SPD + s] = (bf16)0.f;
  base[(size_t)31 * SPD + s] = (bf16)0.f;
}

// ---------------------------------------------------------------------------
// Fused attention: one wave per (qtile16, head, batch).
// scores->LDS via WMMA, softmax over valid 516 keys, P(bf16,LDS) x Vt via WMMA.
// ---------------------------------------------------------------------------
__global__ __launch_bounds__(32) void attn_kernel(
    const bf16* __restrict__ Qbf, const bf16* __restrict__ Kbf,
    const bf16* __restrict__ Vt, bf16* __restrict__ AttnBf) {
  __shared__ float sS[16 * SPD];
  __shared__ bf16  sP[16 * SPD];
  int lane = threadIdx.x;
  int qt = blockIdx.x, h = blockIdx.y, b = blockIdx.z;
  const bf16* Qh = Qbf + ((size_t)b * NH + h) * (size_t)SPD * DHP;
  const bf16* Kh = Kbf + ((size_t)b * NH + h) * (size_t)SPD * DHP;
  const bf16* Vh = Vt  + ((size_t)b * NH + h) * (size_t)DHP * SPD;
  const float scale = 0.18257418583505536f;  // 1/sqrt(30)

  int n  = lane & 15;
  int mo = (lane & 16) ? 8 : 0;

  // scores = Q x K^T  (K-dim = DHP = 32, one WMMA per 16-key tile)
  v16bf aq = load_frag_a(Qh, DHP, qt * 16, 0, lane);
  for (int t = 0; t < SPD / 16; ++t) {
    v16bf bk = load_frag_b(Kh, DHP, t * 16, 0, lane);
    v8f acc = {0.f, 0.f, 0.f, 0.f, 0.f, 0.f, 0.f, 0.f};
    acc = wmma_bf16(aq, bk, acc);
#pragma unroll
    for (int r = 0; r < 8; ++r)
      sS[(mo + r) * SPD + t * 16 + n] = acc[r] * scale;
  }
  __syncthreads();

  // softmax: lane pair (r, r+16) handles row r; columns interleaved mod 2,
  // pairwise combine via shfl_xor(16). Padded keys -> P = 0.
  {
    int half = lane >> 4;            // 0 or 1
    float* row = sS + n * SPD;
    bf16* prow = sP + n * SPD;
    float mx = -3.0e38f;
    for (int j = half; j < SEQV; j += 2) mx = fmaxf(mx, row[j]);
    mx = fmaxf(mx, __shfl_xor(mx, 16, 32));
    float sum = 0.f;
    for (int j = half; j < SEQV; j += 2) {
      float e = __expf(row[j] - mx);
      row[j] = e;
      sum += e;
    }
    sum += __shfl_xor(sum, 16, 32);
    float inv = 1.f / sum;
    for (int j = half; j < SEQV; j += 2) prow[j] = (bf16)(row[j] * inv);
    for (int j = SEQV + half; j < SPD; j += 2) prow[j] = (bf16)0.f;
  }
  __syncthreads();

  // O = P x V  (K-dim = SPD in chunks of 32, N over two 16-wide d tiles)
  for (int dt = 0; dt < 2; ++dt) {
    v8f acc = {0.f, 0.f, 0.f, 0.f, 0.f, 0.f, 0.f, 0.f};
    for (int kc = 0; kc < SPD; kc += 32) {
      v16bf ap = load_frag_a(sP, SPD, 0, kc, lane);
      v16bf bv = load_frag_b(Vh, SPD, dt * 16, kc, lane);
      acc = wmma_bf16(ap, bv, acc);
    }
    int d = dt * 16 + n;
    if (d < DH) {
#pragma unroll
      for (int r = 0; r < 8; ++r) {
        int s = qt * 16 + mo + r;
        AttnBf[((size_t)b * SPD + s) * DPAD + h * DH + d] = (bf16)acc[r];
      }
    }
  }
}

// ---------------------------------------------------------------------------
// x = LayerNorm(x0 + y) over D=240; writes f32 master + padded bf16 copy.
// ---------------------------------------------------------------------------
__global__ __launch_bounds__(32) void resid_ln_kernel(
    const float* __restrict__ X0, const float* __restrict__ Y,
    const float* __restrict__ g, const float* __restrict__ be,
    float* __restrict__ Xout, bf16* __restrict__ Xbf) {
  int row = blockIdx.x, lane = threadIdx.x;
  const float* x0 = X0 + (size_t)row * DM;
  const float* y  = Y  + (size_t)row * DM;
  float v[8];
  float s = 0.f, s2 = 0.f;
  int d0 = lane * 8;
#pragma unroll
  for (int i = 0; i < 8; ++i) {
    int d = d0 + i;
    float t = (d < DM) ? (x0[d] + y[d]) : 0.f;
    v[i] = t; s += t; s2 += t * t;
  }
  s = wave_sum(s); s2 = wave_sum(s2);
  float mu = s / DM;
  float var = s2 / DM - mu * mu;
  float inv = rsqrtf(var + 1e-5f);
#pragma unroll
  for (int i = 0; i < 8; ++i) {
    int d = d0 + i;
    if (d < DM) {
      float o = (v[i] - mu) * inv * g[d] + be[d];
      Xout[(size_t)row * DM + d] = o;
      Xbf[(size_t)row * DPAD + d] = (bf16)o;
    }
  }
}

// relu(LayerNorm(x)) for the MLP head, generic D, bf16 output
__global__ __launch_bounds__(32) void ln_relu_kernel(
    const float* __restrict__ X, const float* __restrict__ g,
    const float* __restrict__ be, bf16* __restrict__ Out, int D) {
  int row = blockIdx.x, lane = threadIdx.x;
  const float* x = X + (size_t)row * D;
  float s = 0.f, s2 = 0.f;
  for (int d = lane; d < D; d += 32) { float t = x[d]; s += t; s2 += t * t; }
  s = wave_sum(s); s2 = wave_sum(s2);
  float mu = s / D;
  float var = s2 / D - mu * mu;
  float inv = rsqrtf(var + 1e-5f);
  for (int d = lane; d < D; d += 32) {
    float o = (x[d] - mu) * inv * g[d] + be[d];
    Out[(size_t)row * D + d] = (bf16)fmaxf(o, 0.f);
  }
}

// fused[B,SPD,16] f32 -> data[B, 516*16] bf16 (drop padded rows)
__global__ void data_pack_kernel(const float* __restrict__ Pf, bf16* __restrict__ Out) {
  int gid = blockIdx.x * blockDim.x + threadIdx.x;
  if (gid >= BB * SEQV * 16) return;
  int j = gid % (SEQV * 16);
  int b = gid / (SEQV * 16);
  int s = j >> 4, d = j & 15;
  Out[(size_t)b * (SEQV * 16) + j] = (bf16)(Pf[((size_t)b * SPD + s) * 16 + d]);
}

// ---------------------------------------------------------------------------
// Host side
// ---------------------------------------------------------------------------
struct ModelParams {
  const float *hand, *head, *chand, *chead, *state, *text, *perc, *htr, *headtr;
  const float *me_img, *me_3d;
  const float *Wq[4], *bq[4], *Wk[4], *bk[4], *Wv[4], *bv[4], *Wo[4], *bo[4];
  const float *W1[4], *b1[4], *W2[4], *b2[4], *g1[4], *be1[4], *g2[4], *be2[4];
  const float *Wproj, *bproj, *Wm1, *bm1, *gm1, *bem1, *Wm2, *bm2, *gm2, *bem2, *Wm3, *bm3;
};

static void map_params(void* const* d_in, const int* in_sizes, int n_in, ModelParams& p) {
  auto F = [&](int i) { return (const float*)d_in[i]; };
  // setup_inputs() insertion order => in_sizes[0] == 32*256*240.
  // jax-sorted pytree order      => in_sizes[0] == 32*256*3 (coords_hand first).
  bool sorted = (n_in > 0 && in_sizes[0] == 32 * 256 * 3);
  if (!sorted) {
    p.hand = F(0); p.head = F(1); p.chand = F(2); p.chead = F(3);
    p.state = F(4); p.text = F(5); p.perc = F(6); p.htr = F(7); p.headtr = F(8);
    p.me_img = F(9); p.me_3d = F(10);
    for (int l = 0; l < 4; ++l) {
      int b = 11 + 16 * l;
      p.Wq[l] = F(b + 0);  p.bq[l] = F(b + 1);  p.Wk[l] = F(b + 2);  p.bk[l] = F(b + 3);
      p.Wv[l] = F(b + 4);  p.bv[l] = F(b + 5);  p.Wo[l] = F(b + 6);  p.bo[l] = F(b + 7);
      p.W1[l] = F(b + 8);  p.b1[l] = F(b + 9);  p.W2[l] = F(b + 10); p.b2[l] = F(b + 11);
      p.g1[l] = F(b + 12); p.be1[l] = F(b + 13); p.g2[l] = F(b + 14); p.be2[l] = F(b + 15);
    }
    p.Wproj = F(75); p.bproj = F(76);
    p.Wm1 = F(77); p.bm1 = F(78); p.gm1 = F(79); p.bem1 = F(80);
    p.Wm2 = F(81); p.bm2 = F(82); p.gm2 = F(83); p.bem2 = F(84);
    p.Wm3 = F(85); p.bm3 = F(86);
  } else {
    p.chand = F(0); p.chead = F(1); p.hand = F(2); p.htr = F(3);
    p.head = F(4); p.headtr = F(5);
    p.Wm1 = F(6); p.Wm2 = F(7); p.Wm3 = F(8); p.Wproj = F(9);
    p.bem1 = F(10); p.bem2 = F(11); p.bm1 = F(12); p.bm2 = F(13); p.bm3 = F(14);
    p.bproj = F(15); p.gm1 = F(16); p.gm2 = F(17);
    for (int l = 0; l < 4; ++l) {
      int b = 18 + 16 * l;  // sorted keys: W1 W2 Wk Wo Wq Wv b1 b2 be1 be2 bk bo bq bv g1 g2
      p.W1[l] = F(b + 0);  p.W2[l] = F(b + 1);  p.Wk[l] = F(b + 2);  p.Wo[l] = F(b + 3);
      p.Wq[l] = F(b + 4);  p.Wv[l] = F(b + 5);  p.b1[l] = F(b + 6);  p.b2[l] = F(b + 7);
      p.be1[l] = F(b + 8); p.be2[l] = F(b + 9); p.bk[l] = F(b + 10); p.bo[l] = F(b + 11);
      p.bq[l] = F(b + 12); p.bv[l] = F(b + 13); p.g1[l] = F(b + 14); p.g2[l] = F(b + 15);
    }
    p.me_3d = F(82); p.me_img = F(83); p.perc = F(84); p.state = F(85); p.text = F(86);
  }
}

extern "C" void kernel_launch(void* const* d_in, const int* in_sizes, int n_in,
                              void* d_out, int out_size, void* d_ws, size_t ws_size,
                              hipStream_t stream) {
  ModelParams mp;
  map_params(d_in, in_sizes, n_in, mp);

  // ---- workspace layout (bump allocator, 256B aligned) ----
  size_t off = 0;
  auto alloc = [&](size_t bytes) {
    size_t o = off;
    off += (bytes + 255) & ~(size_t)255;
    return o;
  };
  char* ws = (char*)d_ws;
  float* Xres   = (float*)(ws + alloc((size_t)BS * DM * 4));
  bf16*  Xbf    = (bf16*) (ws + alloc((size_t)BS * DPAD * 2));
  float* Tf     = (float*)(ws + alloc((size_t)BS * DM * 4));
  bf16*  Qb     = (bf16*) (ws + alloc((size_t)BB * NH * SPD * DHP * 2));
  bf16*  Kb     = (bf16*) (ws + alloc((size_t)BB * NH * SPD * DHP * 2));
  bf16*  Vt     = (bf16*) (ws + alloc((size_t)BB * NH * DHP * SPD * 2));
  bf16*  Att    = (bf16*) (ws + alloc((size_t)BS * DPAD * 2));
  bf16*  Hbf    = (bf16*) (ws + alloc((size_t)BS * FFN * 2));
  float* coords = (float*)(ws + alloc((size_t)BS * 3 * 4));
  float* Pf     = (float*)(ws + alloc((size_t)BS * 16 * 4));
  bf16*  databf = (bf16*) (ws + alloc((size_t)BB * SEQV * 16 * 2));
  float* h1     = (float*)(ws + alloc((size_t)BB * 2048 * 4));
  bf16*  h1b    = (bf16*) (ws + alloc((size_t)BB * 2048 * 2));
  float* h2     = (float*)(ws + alloc((size_t)BB * 512 * 4));
  bf16*  h2b    = (bf16*) (ws + alloc((size_t)BB * 512 * 2));
  // bf16 weight copies (padded K)
  bf16* WqB[4], *WkB[4], *WvB[4], *WoB[4], *W1B[4], *W2B[4];
  for (int l = 0; l < 4; ++l) {
    WqB[l] = (bf16*)(ws + alloc((size_t)DM * DPAD * 2));
    WkB[l] = (bf16*)(ws + alloc((size_t)DM * DPAD * 2));
    WvB[l] = (bf16*)(ws + alloc((size_t)DM * DPAD * 2));
    WoB[l] = (bf16*)(ws + alloc((size_t)DM * DPAD * 2));
    W1B[l] = (bf16*)(ws + alloc((size_t)FFN * DPAD * 2));
    W2B[l] = (bf16*)(ws + alloc((size_t)DM * FFN * 2));
  }
  bf16* WprojB = (bf16*)(ws + alloc((size_t)16 * DPAD * 2));
  bf16* Wm1B   = (bf16*)(ws + alloc((size_t)2048 * (SEQV * 16) * 2));
  bf16* Wm2B   = (bf16*)(ws + alloc((size_t)512 * 2048 * 2));
  bf16* Wm3B   = (bf16*)(ws + alloc((size_t)128 * 512 * 2));
  size_t used = off < ws_size ? off : ws_size;

  // zero padding regions (deterministic; WMMA padded-K lanes contribute 0)
  hipMemsetAsync(d_ws, 0, used, stream);

  // weight conversion to padded bf16
  auto wconv = [&](const float* W, bf16* Wb, int N, int K, int Kpad) {
    int total = N * Kpad;
    wconv_kernel<<<(total + 255) / 256, 256, 0, stream>>>(W, Wb, N, K, Kpad);
  };
  for (int l = 0; l < 4; ++l) {
    wconv(mp.Wq[l], WqB[l], DM, DM, DPAD);
    wconv(mp.Wk[l], WkB[l], DM, DM, DPAD);
    wconv(mp.Wv[l], WvB[l], DM, DM, DPAD);
    wconv(mp.Wo[l], WoB[l], DM, DM, DPAD);
    wconv(mp.W1[l], W1B[l], FFN, DM, DPAD);
    wconv(mp.W2[l], W2B[l], DM, FFN, FFN);
  }
  wconv(mp.Wproj, WprojB, 16, DM, DPAD);
  wconv(mp.Wm1, Wm1B, 2048, SEQV * 16, SEQV * 16);
  wconv(mp.Wm2, Wm2B, 512, 2048, 2048);
  wconv(mp.Wm3, Wm3B, 128, 512, 512);

  assemble_kernel<<<(BS + 255) / 256, 256, 0, stream>>>(
      mp.hand, mp.head, mp.chand, mp.chead, mp.state, mp.text, mp.perc,
      mp.htr, mp.headtr, mp.me_img, mp.me_3d, Xres, Xbf, coords);

  // Layer GEMM grids: 4 waves/block, MREP=4 -> 16 M-tiles per block
  const int GX = MT / 16;  // 68
  dim3 g240(GX, DM / 16);
  dim3 gffn(GX, FFN / 16);
  int packBlocks = (BB * NH * SPD + 255) / 256;

  for (int L = 0; L < 4; ++L) {
    // Q/K/V projections + rotary packs
    gemm_kernel<4, 0, 0><<<g240, 128, 0, stream>>>(Xbf, DPAD, WqB[L], DPAD, DPAD, mp.bq[L], Tf, nullptr, DM);
    rope_pack_kernel<<<packBlocks, 256, 0, stream>>>(Tf, coords, Qb);
    gemm_kernel<4, 0, 0><<<g240, 128, 0, stream>>>(Xbf, DPAD, WkB[L], DPAD, DPAD, mp.bk[L], Tf, nullptr, DM);
    rope_pack_kernel<<<packBlocks, 256, 0, stream>>>(Tf, coords, Kb);
    gemm_kernel<4, 0, 0><<<g240, 128, 0, stream>>>(Xbf, DPAD, WvB[L], DPAD, DPAD, mp.bv[L], Tf, nullptr, DM);
    v_pack_kernel<<<packBlocks, 256, 0, stream>>>(Tf, Vt);

    // fused softmax attention
    attn_kernel<<<dim3(SPD / 16, NH, BB), 32, 0, stream>>>(Qb, Kb, Vt, Att);

    // output projection + residual LN
    gemm_kernel<4, 0, 0><<<g240, 128, 0, stream>>>(Att, DPAD, WoB[L], DPAD, DPAD, mp.bo[L], Tf, nullptr, DM);
    resid_ln_kernel<<<BS, 32, 0, stream>>>(Xres, Tf, mp.g1[L], mp.be1[L], Xres, Xbf);

    // FFN: GELU fused into up-proj epilogue (bf16 out), down-proj, residual LN
    gemm_kernel<4, 1, 1><<<gffn, 128, 0, stream>>>(Xbf, DPAD, W1B[L], DPAD, DPAD, mp.b1[L], nullptr, Hbf, FFN);
    gemm_kernel<4, 0, 0><<<g240, 128, 0, stream>>>(Hbf, FFN, W2B[L], FFN, FFN, mp.b2[L], Tf, nullptr, DM);
    resid_ln_kernel<<<BS, 32, 0, stream>>>(Xres, Tf, mp.g2[L], mp.be2[L], Xres, Xbf);
  }

  // projection to 16 dims, pack to [B, 8256], MLP head
  gemm_kernel<4, 0, 0><<<dim3(GX, 1), 128, 0, stream>>>(Xbf, DPAD, WprojB, DPAD, DPAD, mp.bproj, Pf, nullptr, 16);
  data_pack_kernel<<<(BB * SEQV * 16 + 255) / 256, 256, 0, stream>>>(Pf, databf);

  // MLP head: M = 32 rows = 2 tiles -> MREP=2, single wave per block
  gemm_kernel<2, 0, 0><<<dim3(1, 128), 32, 0, stream>>>(databf, SEQV * 16, Wm1B, SEQV * 16, SEQV * 16, mp.bm1, h1, nullptr, 2048);
  ln_relu_kernel<<<BB, 32, 0, stream>>>(h1, mp.gm1, mp.bem1, h1b, 2048);
  gemm_kernel<2, 0, 0><<<dim3(1, 32), 32, 0, stream>>>(h1b, 2048, Wm2B, 2048, 2048, mp.bm2, h2, nullptr, 512);
  ln_relu_kernel<<<BB, 32, 0, stream>>>(h2, mp.gm2, mp.bem2, h2b, 512);
  gemm_kernel<2, 0, 0><<<dim3(1, 8), 32, 0, stream>>>(h2b, 512, Wm3B, 512, 512, mp.bm3, (float*)d_out, nullptr, 128);
}
